// DeepHedgingLoss_71846212927726
// MI455X (gfx1250) — matrix operand order repaired and checked by reading, not compiled, not run on hardware
//
#include <hip/hip_runtime.h>
#include <hip/hip_bf16.h>
#include <math.h>

typedef __attribute__((ext_vector_type(16))) _Float16 v16h;
typedef __attribute__((ext_vector_type(8)))  float    v8f;
typedef __attribute__((ext_vector_type(8)))  int      v8i;

// v16h (8 dwords) assembled from packed f16 pairs
union V16HU {
    v16h h;
    v8i  i;
};

__device__ __forceinline__ int pkrtz(float a, float b) {
    // single v_cvt_pk_rtz_f16_f32, reinterpreted as one dword
    return __builtin_bit_cast(int, __builtin_amdgcn_cvt_pkrtz(a, b));
}

// lane l <-> lane l^16 exchange on the VALU pipe (v_permlanex16_b32 with
// identity selects: each lane reads the opposite 16-lane row, same position).
__device__ __forceinline__ int xor16_i(int x) {
    unsigned u = (unsigned)x;
    return (int)__builtin_amdgcn_permlanex16(u, u, 0x76543210u, 0xFEDCBA98u, false, false);
}
__device__ __forceinline__ float xor16_f(float x) {
    return __builtin_bit_cast(float, xor16_i(__builtin_bit_cast(int, x)));
}

// --- tanh(z/2) primitive: gfx1250 v_tanh_f32 (confirmed selected), else exp2+rcp ---
#if __has_builtin(__builtin_amdgcn_tanhf)
#define TANHF(x) __builtin_amdgcn_tanhf(x)
#define HAVE_TANH 1
#elif __has_builtin(__builtin_amdgcn_tanh_f32)
#define TANHF(x) __builtin_amdgcn_tanh_f32(x)
#define HAVE_TANH 1
#else
#define HAVE_TANH 0
#endif

// sigmoid jet core: returns sg = sigmoid(z), sp = sigma', om2 = (1 - 2*sg)
__device__ __forceinline__ void sigmoid_core(float z, float& sg, float& sp, float& om2) {
#if HAVE_TANH
    float tn = TANHF(0.5f * z);                 // 1 TRANS
    sg  = fmaf(0.5f, tn, 0.5f);
    sp  = fmaf(-0.25f * tn, tn, 0.25f);         // 0.25*(1 - tn^2)
    om2 = -tn;                                  // 1 - 2*sigmoid
#else
    float e = __builtin_amdgcn_exp2f(z * -1.4426950408889634f);  // exp(-z)
    sg  = __builtin_amdgcn_rcpf(1.0f + e);
    sp  = sg * (1.0f - sg);
    om2 = fmaf(-2.0f, sg, 1.0f);
#endif
}

#define NPATH        8192
#define NSTEPS       128
#define WIDTH        32
#define NHID         3
#define PATHS_WAVE   16
#define WAVES_WG     4
#define PATHS_WG     (PATHS_WAVE * WAVES_WG)   // 64
#define THREADS_WG   (32 * WAVES_WG)           // 128
#define LDS_STRIDE   132                       // pad -> conflict-free per-step column reads

// silu forward-mode jet: value, d/dt, d/ds, d2/ds2 propagation through silu(z)
__device__ __forceinline__ void silu_jet(float z, float zt, float zs, float zss,
                                         float& av, float& at, float& as_, float& ass) {
    float sg, sp, om2;
    sigmoid_core(z, sg, sp, om2);
    float d1 = fmaf(z, sp, sg);                 // silu'
    float d2 = sp * fmaf(om2, z, 2.0f);         // silu'' = sp*(2 + z*(1-2sg))
    av  = z * sg;
    at  = d1 * zt;
    as_ = d1 * zs;
    ass = fmaf(d2 * zs, zs, d1 * zss);
}

// layer-0 variant: zss == 0, so drop the d1*zss term (compiler can't fold x*0)
__device__ __forceinline__ void silu_jet0(float z, float zt, float zs,
                                          float& av, float& at, float& as_, float& ass) {
    float sg, sp, om2;
    sigmoid_core(z, sg, sp, om2);
    float d1 = fmaf(z, sp, sg);                 // silu'
    float d2 = sp * fmaf(om2, z, 2.0f);         // silu''
    av  = z * sg;
    at  = d1 * zt;
    as_ = d1 * zs;
    ass = (d2 * zs) * zs;
}

// Repack two f32 C/D-layout row-tiles (rows 0-15 = t0, rows 16-31 = t1) of one
// channel into the f16 B-operand layout (32x16, K = rows, N = paths).
// Pack to dwords first, then exchange dwords across half-waves.
__device__ __forceinline__ v16h pack_B(const float* t0, const float* t1, bool hi) {
    int X[4], Y[4];
#pragma unroll
    for (int d = 0; d < 4; ++d) {
        X[d] = pkrtz(t0[2 * d], t0[2 * d + 1]);   // tile0 row-pair dword
        Y[d] = pkrtz(t1[2 * d], t1[2 * d + 1]);   // tile1 row-pair dword
    }
    V16HU b;
#pragma unroll
    for (int d = 0; d < 4; ++d) {
#if __has_builtin(__builtin_amdgcn_permlane16_swap)
        // (Y', X') = swap(Y, X): Y'(lo)=X(hi-partner) -> Bhi ; X'(hi)=Y(lo-partner) -> Blo
        auto r = __builtin_amdgcn_permlane16_swap((unsigned)Y[d], (unsigned)X[d], false, false);
        b.i[d]     = (int)r[1];          // Blo: lo lanes own X, hi lanes partner's Y
        b.i[4 + d] = (int)r[0];          // Bhi: lo lanes partner's X, hi lanes own Y
#else
        int send = hi ? X[d] : Y[d];
        int recv = xor16_i(send);
        b.i[d]     = hi ? recv : X[d];   // K(2d,2d+1)   local
        b.i[4 + d] = hi ? Y[d] : recv;   // K(8+2d,9+2d) local
#endif
    }
    return b.h;
}

__device__ __forceinline__ v8f wmma_f16(v16h a, v16h b, v8f c) {
    return __builtin_amdgcn_wmma_f32_16x16x32_f16(false, a, false, b, (short)0, c, false, false);
}

__global__ __launch_bounds__(THREADS_WG)
void deep_hedge_kernel(const float* __restrict__ rnorm,
                       const float* __restrict__ W0, const float* __restrict__ b0,
                       const float* __restrict__ Wh, const float* __restrict__ bh,
                       const float* __restrict__ Wf, const float* __restrict__ bf,
                       float* __restrict__ out) {
    __shared__ float lds_rn[PATHS_WG * LDS_STRIDE];

    const int  tid  = threadIdx.x;
    const int  wave = tid >> 5;
    const int  lane = tid & 31;
    const bool hi   = (lane & 16) != 0;   // which half-wave
    const int  col  = lane & 15;          // path slot within wave tile
    const int  wgP0 = blockIdx.x * PATHS_WG;
    const int  path = wgP0 + wave * PATHS_WAVE + col;

    // ---- stage rnorm tile [64 paths x 128 steps] into LDS, coalesced ----
    for (int i = tid; i < PATHS_WG * NSTEPS; i += THREADS_WG) {
        int p  = i >> 7;       // / 128
        int st = i & 127;
        lds_rn[p * LDS_STRIDE + st] = rnorm[(wgP0 + p) * NSTEPS + st];
    }
    __syncthreads();

    // ---- preload hidden-layer weights as WMMA A operands (loop-invariant) ----
    // A packing used here (contiguous k, same order as pack_B): lane l<16 rows M=l; vgpr v -> K=2v(+1)
    //                     lane l>=16 same M, K offset +16
    v16h A[NHID][2];
#pragma unroll
    for (int L = 0; L < NHID; ++L) {
#pragma unroll
        for (int T = 0; T < 2; ++T) {
            V16HU a;
#pragma unroll
            for (int v = 0; v < 8; ++v) {
                int k = 2 * v + (hi ? 16 : 0);
                const float* src = Wh + L * WIDTH * WIDTH + (T * 16 + col) * WIDTH + k;
                a.i[v] = pkrtz(src[0], src[1]);
            }
            A[L][T] = a.h;
        }
    }

    // per-lane row index for accumulator layout slot (T, v): row = T*16 + v + 8*hi
    float BH[NHID][2][8];
#pragma unroll
    for (int L = 0; L < NHID; ++L)
#pragma unroll
        for (int T = 0; T < 2; ++T)
#pragma unroll
            for (int v = 0; v < 8; ++v)
                BH[L][T][v] = bh[L * WIDTH + T * 16 + v + (hi ? 8 : 0)];

    float W0a[2][8], W0b[2][8], B0r[2][8], WFr[2][8];
#pragma unroll
    for (int T = 0; T < 2; ++T)
#pragma unroll
        for (int v = 0; v < 8; ++v) {
            int row = T * 16 + v + (hi ? 8 : 0);
            W0a[T][v] = W0[row * 2 + 0];
            W0b[T][v] = W0[row * 2 + 1];
            B0r[T][v] = b0[row];
            WFr[T][v] = Wf[row];
        }
    const float bf0 = bf[0];

    // ---- SDE state (duplicated in both half-wave lanes of a path) ----
    const float dt  = 1.0f / (float)NSTEPS;
    const float sdt = sqrtf(dt);
    float s  = 1.0f;   // asset price
    float vg = 0.0f;   // hedging gains

    for (int st = 0; st < NSTEPS; ++st) {
        const float t  = dt * (float)st;
        const float dw = sdt * lds_rn[(wave * PATHS_WAVE + col) * LDS_STRIDE + st];

        // ---- layer 0 (2 -> 32) + silu jet, directly in accumulator layout ----
        float hv[2][8], ht[2][8], hs[2][8], hss[2][8];
#pragma unroll
        for (int T = 0; T < 2; ++T)
#pragma unroll
            for (int v = 0; v < 8; ++v) {
                float z  = fmaf(W0a[T][v], t, fmaf(W0b[T][v], s, B0r[T][v]));
                silu_jet0(z, W0a[T][v], W0b[T][v],
                          hv[T][v], ht[T][v], hs[T][v], hss[T][v]);
            }

        // ---- hidden layers: 2 row-tiles x 4 channels = 8 WMMAs each ----
#pragma unroll
        for (int L = 0; L < NHID; ++L) {
            v16h Bv  = pack_B(hv[0],  hv[1],  hi);
            v16h Bt  = pack_B(ht[0],  ht[1],  hi);
            v16h Bs  = pack_B(hs[0],  hs[1],  hi);
            v16h Bss = pack_B(hss[0], hss[1], hi);

            v8f cb0, cb1;
            const v8f cz = {0.f, 0.f, 0.f, 0.f, 0.f, 0.f, 0.f, 0.f};
#pragma unroll
            for (int v = 0; v < 8; ++v) { cb0[v] = BH[L][0][v]; cb1[v] = BH[L][1][v]; }

            v8f zv0  = wmma_f16(A[L][0], Bv,  cb0);
            v8f zv1  = wmma_f16(A[L][1], Bv,  cb1);
            v8f zt0  = wmma_f16(A[L][0], Bt,  cz);
            v8f zt1  = wmma_f16(A[L][1], Bt,  cz);
            v8f zs0  = wmma_f16(A[L][0], Bs,  cz);
            v8f zs1  = wmma_f16(A[L][1], Bs,  cz);
            v8f zss0 = wmma_f16(A[L][0], Bss, cz);
            v8f zss1 = wmma_f16(A[L][1], Bss, cz);

#pragma unroll
            for (int v = 0; v < 8; ++v) {
                silu_jet(zv0[v], zt0[v], zs0[v], zss0[v],
                         hv[0][v], ht[0][v], hs[0][v], hss[0][v]);
                silu_jet(zv1[v], zt1[v], zs1[v], zss1[v],
                         hv[1][v], ht[1][v], hs[1][v], hss[1][v]);
            }
        }

        // ---- head: u = Wf @ h (+ bf on value channel), reduce across half-waves ----
        float uv = 0.f, ut = 0.f, us = 0.f, uss = 0.f;
#pragma unroll
        for (int T = 0; T < 2; ++T)
#pragma unroll
            for (int v = 0; v < 8; ++v) {
                float w = WFr[T][v];
                uv  = fmaf(w, hv[T][v],  uv);
                ut  = fmaf(w, ht[T][v],  ut);
                us  = fmaf(w, hs[T][v],  us);
                uss = fmaf(w, hss[T][v], uss);
            }
        uv  += xor16_f(uv);
        ut  += xor16_f(ut);
        us  += xor16_f(us);
        uss += xor16_f(uss);
        uv  += bf0;

        // sigmoid jet for the head
        float p, pp, om2;
        sigmoid_core(uv, p, pp, om2);
        (void)p;
        float dhdt  = pp * ut;
        float dhds  = pp * us;
        float dhdss = fmaf(pp * om2 * us, us, pp * uss);   // pp*(1-2p)*us^2 + pp*uss

        // ---- Ito-Milstein step (MU = SIGMA = 1) ----
        float dw2 = dw * dw - dt;
        float fV  = dhdt + dhds * s + 0.5f * dhdss * s * s;   // drift of v
        float gV  = dhds * s;                                 // diffusion of v
        float mV  = (dhdss * s + dhds) * s;                   // (Dg.g)_v
        float sn  = s + s * dt + s * dw + 0.5f * s * dw2;     // MU = SIGMA = 1
        vg        = vg + fV * dt + gV * dw + 0.5f * mV * dw2;
        s = sn;
    }

    if (!hi) {                      // lanes 0-15 hold the canonical copy
        out[path * 2 + 0] = s;
        out[path * 2 + 1] = vg;
    }
}

extern "C" void kernel_launch(void* const* d_in, const int* in_sizes, int n_in,
                              void* d_out, int out_size, void* d_ws, size_t ws_size,
                              hipStream_t stream) {
    const float* rnorm = (const float*)d_in[0];
    const float* W0    = (const float*)d_in[1];
    const float* b0    = (const float*)d_in[2];
    const float* Wh    = (const float*)d_in[3];
    const float* bh    = (const float*)d_in[4];
    const float* Wf    = (const float*)d_in[5];
    const float* bf    = (const float*)d_in[6];
    float* out = (float*)d_out;

    dim3 grid(NPATH / PATHS_WG);    // 128 workgroups
    dim3 block(THREADS_WG);         // 128 threads = 4 wave32
    deep_hedge_kernel<<<grid, block, 0, stream>>>(rnorm, W0, b0, Wh, bh, Wf, bf, out);
}
